// QuanConv_7112465842389
// MI455X (gfx1250) — matrix operand order repaired and checked by reading, not compiled, not run on hardware
//
#include <hip/hip_runtime.h>
#include <hip/hip_bf16.h>

typedef _Float16 h4   __attribute__((ext_vector_type(4)));
typedef _Float16 h8   __attribute__((ext_vector_type(8)));
typedef _Float16 v16h __attribute__((ext_vector_type(16)));
typedef float    v8f  __attribute__((ext_vector_type(8)));
typedef unsigned int u32x4 __attribute__((ext_vector_type(4)));
typedef int          i32x4 __attribute__((ext_vector_type(4)));
typedef int          i32x8 __attribute__((ext_vector_type(8)));

#define CIN   256
#define COUT  256
#define HW    56
#define NPIX  (HW*HW)          // 3136
#define NTILE 8
#define TILE_C (COUT/NTILE)    // 32
#define WELEMS (COUT*CIN*9)    // 589824
#define LROW  40               // padded LDS row stride in halves (80 B)

#if __has_builtin(__builtin_amdgcn_tensor_load_to_lds)
#define HAVE_TDM 1
#else
#define HAVE_TDM 0
#endif

// ws layout: [0]=uint maxv bits, [1..8]=sel per tile, wq fp16 at byte 64 (9*256*256 halves)

// ---------------- gate + init ----------------
__global__ void gate_kernel(const float* __restrict__ logits,
                            const float* __restrict__ gumbel,
                            unsigned* __restrict__ ws) {
    int t = threadIdx.x;
    if (t < NTILE) {
        float best = logits[t*4] + gumbel[t*4];
        int bi = 0;
        #pragma unroll
        for (int j = 1; j < 4; ++j) {
            float v = logits[t*4 + j] + gumbel[t*4 + j];
            if (v > best) { best = v; bi = j; }
        }
        ((int*)ws)[1 + t] = bi;
    }
    if (t == NTILE) ws[0] = 0u;   // maxv accumulator (floats >=0 -> uint order ok)
}

// ---------------- max |tanh(w)| ----------------
__global__ void wmax_kernel(const float* __restrict__ w, unsigned* __restrict__ ws) {
    __shared__ float red[256];
    float m = 0.f;
    for (int i = blockIdx.x * 256 + threadIdx.x; i < WELEMS; i += gridDim.x * 256)
        m = fmaxf(m, fabsf(tanhf(w[i])));
    red[threadIdx.x] = m;
    __syncthreads();
    for (int s = 128; s > 0; s >>= 1) {
        if (threadIdx.x < s) red[threadIdx.x] = fmaxf(red[threadIdx.x], red[threadIdx.x + s]);
        __syncthreads();
    }
    if (threadIdx.x == 0) atomicMax(ws, __float_as_uint(red[0]));
}

// ------ per-channel weight quantization -> fp16 tensor [khw][co][ci] ------
__global__ void wquant_kernel(const float* __restrict__ w,
                              const unsigned* __restrict__ ws,
                              _Float16* __restrict__ wq) {
    int d = blockIdx.x * 256 + threadIdx.x;       // dst index
    if (d >= WELEMS) return;
    int khw = d >> 16;                            // /65536
    int r   = d & 0xFFFF;
    int co  = r >> 8;
    int ci  = r & 0xFF;
    float v = w[(co*CIN + ci)*9 + khw];           // OIHW source
    int nb  = ((const int*)ws)[1 + (co / TILE_C)];
    float o;
    if (nb == 3) {                                // 32-bit: passthrough
        o = v;
    } else {
        const float scl = (nb == 0) ? 15.f : (nb == 1) ? 255.f : 65535.f;
        float maxv = __uint_as_float(ws[0]);
        float u = tanhf(v) / (2.f * maxv) + 0.5f;
        float q = rintf(u * scl) * (1.f / scl);
        o = 2.f * q - 1.f;
    }
    wq[d] = (_Float16)o;
}

// ---------------- implicit-GEMM conv with WMMA + TDM ----------------
// grid: (49 = 7x7 2-D pixel tiles, 8 channel-tiles, 32 batch), block 256 = 8 wave32
__global__ void __launch_bounds__(256)
conv_kernel(const float* __restrict__ x,
            const _Float16* __restrict__ wq,
            const float* __restrict__ bias,
            const unsigned* __restrict__ ws,
            float* __restrict__ out) {
    // weights: 9 taps x 32 co, padded rows of 40 halves (TDM pad keeps same stride)
    __shared__ __align__(16) _Float16 lws[9 * TILE_C * LROW];   // 23040 B
    // activations: 10x10 halo tile x 32 ci, padded rows
    __shared__ __align__(16) _Float16 la[10 * 10 * LROW];       //  8000 B

    const int tid  = threadIdx.x;
    const int lane = tid & 31;
    const int wid  = tid >> 5;            // 0..7
    const int wm   = wid & 1;             // co half (16 rows)
    const int wn   = wid >> 1;            // pixel quarter (2 rows x 8 cols)

    const int pblk = blockIdx.x;          // 0..48
    const int tile = blockIdx.y;          // 0..7
    const int n    = blockIdx.z;          // 0..31
    const int co0  = tile * TILE_C;
    const int r0   = (pblk / 7) * 8;      // output row origin
    const int c0   = (pblk % 7) * 8;      // output col origin

    const int nb = ((const int*)ws)[1 + tile];
    // nb==3 (fp32): use a grid far finer than fp16 ulp -> identity after fp16 cvt
    const float ascale = (nb == 0) ? 15.f : (nb == 1) ? 255.f :
                         (nb == 2) ? 65535.f : 16777216.f;
    const float ainv   = (nb == 0) ? (1.f/15.f) : (nb == 1) ? (1.f/255.f) :
                         (nb == 2) ? (1.f/65535.f) : (1.f/16777216.f);

    // fragment geometry (ISA 7.12.2 layouts, wave32)
    const int mrow = (lane & 15) + 16 * wm;     // A row (co local)
    const int akb  = (lane >> 4) * 8;           // A K base (0/8)
    const int nn   = lane & 15;
    const int pr   = 2 * wn + (nn >> 3);        // output pixel row in 8x8 tile
    const int pc   = nn & 7;                    // output pixel col
    const int bkb  = (lane >> 4) * 8;           // B K base (0/8)

    v8f acc[2] = {{}, {}};

    for (int ci0 = 0; ci0 < CIN; ci0 += 32) {
        // ---- weights for ALL 9 taps of this ci-chunk: one 3-D TDM tile ----
        if (wid == 0) {
#if HAVE_TDM
            const unsigned lds_dst = (unsigned)(uintptr_t)(void*)lws;
            const unsigned long long ga =
                (unsigned long long)(uintptr_t)(wq + (size_t)(co0 * CIN + ci0));
            // D# group0: count=1, lds_addr, global_addr, type=2
            u32x4 g0 = { 1u, lds_dst, (unsigned)ga,
                         (unsigned)(ga >> 32) | (2u << 30) };
            // D# group1: data_size=2B, pad_enable, pad every 16 dwords by 4 dwords,
            // tensor 256(ci) x 256(co) [strides 1, 256], tile 32 x 32 x 9
            i32x8 g1 = { (int)((1u << 16) | (1u << 20) | (3u << 22) | (3u << 25)),
                         (int)(256u << 16),          // tensor_dim0=256 (low16 @bit48)
                         (int)(256u << 16),          // dim0 hi=0 | tensor_dim1=256 lo16
                         (int)(32u  << 16),          // dim1 hi=0 | tile_dim0=32
                         (int)(32u | (9u << 16)),    // tile_dim1=32 | tile_dim2=9
                         256,                        // tensor_dim0_stride lo32
                         0,                          // stride0 hi | stride1 lo16 (65536&0xFFFF=0)
                         1 };                        // stride1 bits[47:16] = 65536>>16
            i32x4 g2 = { 9, 0, 0, 0 };               // tensor_dim2=9
            i32x4 g3 = { 0, 0, 0, 0 };
#if defined(__clang_major__) && (__clang_major__ >= 23)
            i32x8 gz = { 0, 0, 0, 0, 0, 0, 0, 0 };
            __builtin_amdgcn_tensor_load_to_lds(g0, g1, g2, g3, gz, 0);
#else
            __builtin_amdgcn_tensor_load_to_lds(g0, g1, g2, g3, 0);
#endif
            __builtin_amdgcn_s_wait_tensorcnt((short)0);
#else
            ; // fallback handled below by all threads
#endif
        }
#if !HAVE_TDM
        // manual fallback: 9216 halves, h4 per thread-iteration
        #pragma unroll
        for (int r = 0; r < 9; ++r) {
            const int f   = (tid + r * 256) * 4;    // flat half index
            const int ci  = f & 31;
            const int row = f >> 5;                 // 0..287 = khw*32+co
            h4 wv = *reinterpret_cast<const h4*>(
                wq + ((size_t)(row >> 5) << 16) + (co0 + (row & 31)) * CIN + ci0 + ci);
            *reinterpret_cast<h4*>(&lws[row * LROW + ci]) = wv;
        }
#endif
        // ---- stage + quantize 10x10 halo x 32ci activations (once per chunk) ----
        for (int i = tid; i < 100 * 32; i += 256) {
            const int px = i % 100;                 // halo-linear pixel
            const int ci = i / 100;
            const int h  = r0 + (px / 10) - 1;
            const int w  = c0 + (px % 10) - 1;
            const bool inb = ((unsigned)h < (unsigned)HW) & ((unsigned)w < (unsigned)HW);
            const int hc = min(max(h, 0), HW - 1);
            const int wc = min(max(w, 0), HW - 1);
            float v = x[((size_t)(n * CIN + ci0 + ci)) * NPIX + hc * HW + wc];
            v = inb ? v : 0.f;
            v = fminf(fmaxf(v, 0.f), 1.f);          // clip(x,0,1)
            v = rintf(v * ascale) * ainv;           // RNE to DoReFa grid
            la[px * LROW + ci] = (_Float16)v;
            if (ci0 + 32 < CIN && i < 256)          // gfx1250 global_prefetch_b8
                __builtin_prefetch(&x[((size_t)(n * CIN + ci0 + 32 + ci)) * NPIX + hc * HW + wc], 0, 1);
        }
        __syncthreads();

        // ---- 9 WMMAs back-to-back (one per tap), 2 interleaved acc chains ----
        #pragma unroll
        for (int tap = 0; tap < 9; ++tap) {
            const int kh = tap / 3, kw = tap % 3;
            const _Float16* ap = lws + ((tap * TILE_C + mrow) * LROW + akb);
            const _Float16* bp = la  + (((pr + kh) * 10 + (pc + kw)) * LROW + bkb);
            h8 a0 = *reinterpret_cast<const h8*>(ap);
            h8 a1 = *reinterpret_cast<const h8*>(ap + 16);
            h8 b0 = *reinterpret_cast<const h8*>(bp);
            h8 b1 = *reinterpret_cast<const h8*>(bp + 16);
            v16h a, b;
            #pragma unroll
            for (int i = 0; i < 8; ++i) {
                a[i] = a0[i]; a[8 + i] = a1[i];
                b[i] = b0[i]; b[8 + i] = b1[i];
            }
            acc[tap & 1] = __builtin_amdgcn_wmma_f32_16x16x32_f16(
                               false, a, false, b, (short)0, acc[tap & 1], false, false);
        }
        __syncthreads();   // before LDS tiles are overwritten
    }

    // ---- store per C/D layout: lane<16 -> rows 0..7, lane>=16 -> rows 8..15 ----
    v8f sum = acc[0] + acc[1];
    const int rbase = (lane >> 4) * 8;
    const int px = (r0 + pr) * HW + (c0 + pc);
    #pragma unroll
    for (int r = 0; r < 8; ++r) {
        const int co = co0 + 16 * wm + rbase + r;
        out[((size_t)(n * COUT + co)) * NPIX + px] = sum[r] + bias[co];
    }
}

extern "C" void kernel_launch(void* const* d_in, const int* in_sizes, int n_in,
                              void* d_out, int out_size, void* d_ws, size_t ws_size,
                              hipStream_t stream) {
    (void)in_sizes; (void)n_in; (void)out_size; (void)ws_size;
    const float* x      = (const float*)d_in[0];
    const float* weight = (const float*)d_in[1];
    const float* bias   = (const float*)d_in[2];
    const float* logits = (const float*)d_in[3];
    const float* gumbel = (const float*)d_in[4];
    float* out = (float*)d_out;

    unsigned*  ws = (unsigned*)d_ws;
    _Float16*  wq = (_Float16*)((char*)d_ws + 64);   // 9*256*256 halves = 1.18 MB

    gate_kernel<<<1, 32, 0, stream>>>(logits, gumbel, ws);
    wmax_kernel<<<576, 256, 0, stream>>>(weight, ws);
    wquant_kernel<<<(WELEMS + 255) / 256, 256, 0, stream>>>(weight, ws, wq);
    conv_kernel<<<dim3(49, NTILE, 32), 256, 0, stream>>>(x, wq, bias, ws, out);
}
